// Net2_77197742178636
// MI455X (gfx1250) — matrix operand order repaired and checked by reading, not compiled, not run on hardware
//
#include <hip/hip_runtime.h>
#include <hip/hip_bf16.h>

typedef __attribute__((ext_vector_type(16))) __bf16 v16bf;
typedef __attribute__((ext_vector_type(8)))  float  v8f;

#define NN   65536      // nodes
#define HH   128        // hidden = feature dim
#define GG   32         // groups
#define SEGS 2048       // nodes per group
#define NCLS 40
#define NCLSP 48        // padded to N-tile multiple

// ---------------- prep kernels ----------------

// Wdyn[g][h][f] (B^T layout, bf16) = relu(w0[f]*E_meta[g,h] + b0[f,h])
__global__ __launch_bounds__(256) void prep_wdyn_kernel(
    const float* __restrict__ w0, const float* __restrict__ Em,
    const float* __restrict__ b0, __bf16* __restrict__ wdyn) {
  int idx = blockIdx.x * 256 + threadIdx.x;          // < G*128*128
  int g = idx >> 14, rem = idx & 16383;
  int n = rem >> 7, k = rem & 127;                    // n = h-index, k = f-index
  float v = w0[k] * Em[(g << 7) + n] + b0[(k << 7) + n];
  wdyn[idx] = (__bf16)(v > 0.f ? v : 0.f);
}

// conv_w[l] transposed to [N][K] bf16
__global__ __launch_bounds__(256) void prep_convwT_kernel(
    const float* __restrict__ cw, __bf16* __restrict__ cwt) {
  int idx = blockIdx.x * 256 + threadIdx.x;          // < 3*128*128
  int l = idx >> 14, rem = idx & 16383;
  int n = rem >> 7, k = rem & 127;
  cwt[idx] = (__bf16)cw[(l << 14) + (k << 7) + n];
}

// lt1_w [40,128] is already [N][K]; pad N to 48
__global__ __launch_bounds__(256) void prep_lt1_kernel(
    const float* __restrict__ l1w, __bf16* __restrict__ l1p) {
  int idx = blockIdx.x * 256 + threadIdx.x;          // < 48*128
  int n = idx >> 7, k = idx & 127;
  l1p[idx] = (__bf16)(n < NCLS ? l1w[n * HH + k] : 0.f);
}

// x -> bf16, 4 elements per thread
__global__ __launch_bounds__(256) void cvt_x_kernel(
    const float* __restrict__ x, __bf16* __restrict__ xb) {
  int idx = blockIdx.x * 256 + threadIdx.x;          // < N*32
  float4 v = ((const float4*)x)[idx];
  alignas(8) __bf16 o[4] = {(__bf16)v.x, (__bf16)v.y, (__bf16)v.z, (__bf16)v.w};
  ((uint2*)xb)[idx] = *(const uint2*)o;
}

__global__ __launch_bounds__(256) void fill4_kernel(float* __restrict__ p,
                                                    float val, int n4) {
  int idx = blockIdx.x * 256 + threadIdx.x;
  if (idx < n4) ((float4*)p)[idx] = make_float4(val, val, val, val);
}

__global__ __launch_bounds__(256) void degree_kernel(
    const int* __restrict__ dst, float* __restrict__ deg, int E) {
  int e = blockIdx.x * 256 + threadIdx.x;
  if (e < E)
    __hip_atomic_fetch_add(&deg[dst[e]], 1.f, __ATOMIC_RELAXED,
                           __HIP_MEMORY_SCOPE_AGENT);
}

__global__ __launch_bounds__(256) void rsqrt_kernel(float* __restrict__ deg) {
  int i = blockIdx.x * 256 + threadIdx.x;            // deg >= 1 (self loops)
  deg[i] = rsqrtf(deg[i]);
}

// ---------------- WMMA GEMM: [M x 128] (bf16) @ Bt[Ntiles*16][128] ----------------
// Block = 256 thr = 8 waves; wave w owns rows [blk*128 + w*16, +16), all NT N-tiles.
template <int NT, bool OUTBF16>
__global__ __launch_bounds__(256) void gemm_bf16_kernel(
    const __bf16* __restrict__ A, const __bf16* __restrict__ Bt,
    float* __restrict__ outF, __bf16* __restrict__ outB,
    int ldout, int rowsPerB) {
  const int wave = threadIdx.x >> 5;
  const int lane = threadIdx.x & 31;
  const int m = lane & 15;
  const int gsel = lane >> 4;                       // lane group 0/1
  const int rowBase = blockIdx.x * 128 + wave * 16;
  const __bf16* Bsel =
      Bt + (size_t)((blockIdx.x * 128) / rowsPerB) * (NT * 16) * HH;

  v8f zero = {0.f, 0.f, 0.f, 0.f, 0.f, 0.f, 0.f, 0.f};
  v8f acc[NT];
#pragma unroll
  for (int t = 0; t < NT; ++t) acc[t] = zero;

  const __bf16* arow = A + (size_t)(rowBase + m) * HH;

#pragma unroll
  for (int kk = 0; kk < HH; kk += 32) {
    // A fragment 16x32: lanes 0-15 K=[kk..kk+7],[kk+16..kk+23]; lanes 16-31 +8
    v16bf a;
    ((uint4*)&a)[0] = *(const uint4*)(arow + kk + gsel * 8);
    ((uint4*)&a)[1] = *(const uint4*)(arow + kk + 16 + gsel * 8);
#pragma unroll
    for (int t = 0; t < NT; ++t) {
      // B fragment 32x16 from Bt[n][k]: lane n = t*16+m, K = [kk+gsel*8..+7],[kk+16+gsel*8..+7]
      const __bf16* bcol = Bsel + (size_t)(t * 16 + m) * HH + kk + gsel * 8;
      v16bf b;
      ((uint4*)&b)[0] = *(const uint4*)(bcol);
      ((uint4*)&b)[1] = *(const uint4*)(bcol + 16);
      acc[t] = __builtin_amdgcn_wmma_f32_16x16x32_bf16(
          false, a, false, b, (short)0, acc[t], false, false);
    }
  }

  // C/D layout: VGPR v -> row = rowBase + gsel*8 + v, col = t*16 + (lane%16)
#pragma unroll
  for (int t = 0; t < NT; ++t) {
#pragma unroll
    for (int v = 0; v < 8; ++v) {
      size_t r = (size_t)(rowBase + gsel * 8 + v);
      int c = t * 16 + m;
      if (OUTBF16)
        outB[r * ldout + c] = (__bf16)acc[t][v];
      else
        outF[r * ldout + c] = acc[t][v];
    }
  }
}

// ---------------- edge scatter: one wave per edge ----------------
__global__ __launch_bounds__(256) void scatter_kernel(
    const int* __restrict__ src, const int* __restrict__ dst,
    const float* __restrict__ dis, const float* __restrict__ hw,
    float* __restrict__ acc, int E) {
  int tid = blockIdx.x * 256 + threadIdx.x;
  int e = tid >> 5, lane = tid & 31;
  if (e >= E) return;
  int s = src[e], d = dst[e];
  float nrm = dis[s] * dis[d];
  const float* hs = hw + (size_t)s * HH + lane;
  float* ad = acc + (size_t)d * HH + lane;
#pragma unroll
  for (int j = 0; j < 4; ++j)
    __hip_atomic_fetch_add(ad + 32 * j, hs[32 * j] * nrm, __ATOMIC_RELAXED,
                           __HIP_MEMORY_SCOPE_AGENT);
}

// h = relu(acc + self_loop + bias) -> bf16 for next GEMM
__global__ __launch_bounds__(256) void finalize_kernel(
    const float* __restrict__ acc, const float* __restrict__ hw,
    const float* __restrict__ dis, const float* __restrict__ cb,
    __bf16* __restrict__ hb) {
  int idx = blockIdx.x * 256 + threadIdx.x;          // < N*128
  int i = idx >> 7, c = idx & 127;
  float d = dis[i];
  float v = acc[idx] + hw[idx] * d * d + cb[c];
  hb[idx] = (__bf16)(v > 0.f ? v : 0.f);
}

__global__ __launch_bounds__(256) void logsoftmax_kernel(
    const float* __restrict__ logits /* ld=48 */,
    const float* __restrict__ l1b, float* __restrict__ out) {
  int n = blockIdx.x * 256 + threadIdx.x;            // < N
  const float* lg = logits + (size_t)n * NCLSP;
  float z[NCLS];
  float mx = -3.4e38f;
#pragma unroll
  for (int c = 0; c < NCLS; ++c) {
    z[c] = lg[c] + l1b[c];
    mx = fmaxf(mx, z[c]);
  }
  float s = 0.f;
#pragma unroll
  for (int c = 0; c < NCLS; ++c) s += __expf(z[c] - mx);
  float ls = __logf(s);
  float* o = out + (size_t)n * NCLS;
#pragma unroll
  for (int c = 0; c < NCLS; ++c) o[c] = z[c] - mx - ls;
}

// ---------------- host ----------------
extern "C" void kernel_launch(void* const* d_in, const int* in_sizes, int n_in,
                              void* d_out, int out_size, void* d_ws,
                              size_t ws_size, hipStream_t stream) {
  const float* x  = (const float*)d_in[0];
  const int*   ei = (const int*)d_in[1];
  const float* Em = (const float*)d_in[2];
  const float* w0 = (const float*)d_in[4];
  const float* b0 = (const float*)d_in[5];
  const float* cw = (const float*)d_in[6];
  const float* cb = (const float*)d_in[7];
  const float* l1w = (const float*)d_in[8];
  const float* l1b = (const float*)d_in[9];
  float* out = (float*)d_out;
  const int E = in_sizes[1] / 2;
  const int* src = ei;
  const int* dst = ei + E;

  // workspace carve-out (256B aligned slices)
  char* ws = (char*)d_ws;
  size_t off = 0;
  auto carve = [&](size_t bytes) {
    void* p = ws + off;
    off = (off + bytes + 255) & ~(size_t)255;
    return p;
  };
  float*  deg  = (float*)carve((size_t)NN * 4);            // -> deg_inv_sqrt
  __bf16* xb   = (__bf16*)carve((size_t)NN * HH * 2);
  __bf16* hb   = (__bf16*)carve((size_t)NN * HH * 2);
  float*  hw   = (float*)carve((size_t)NN * HH * 4);       // reused for logits
  float*  accb = (float*)carve((size_t)NN * HH * 4);
  __bf16* wdyn = (__bf16*)carve((size_t)GG * HH * HH * 2);
  __bf16* cwt  = (__bf16*)carve((size_t)3 * HH * HH * 2);
  __bf16* l1p  = (__bf16*)carve((size_t)NCLSP * HH * 2);

  // ---- prep ----
  prep_wdyn_kernel<<<GG * HH * HH / 256, 256, 0, stream>>>(w0, Em, b0, wdyn);
  prep_convwT_kernel<<<3 * HH * HH / 256, 256, 0, stream>>>(cw, cwt);
  prep_lt1_kernel<<<NCLSP * HH / 256, 256, 0, stream>>>(l1w, l1p);
  cvt_x_kernel<<<NN * 32 / 256, 256, 0, stream>>>(x, xb);

  // ---- degree (self loops via init=1) -> deg_inv_sqrt in place ----
  fill4_kernel<<<(NN / 4 + 255) / 256, 256, 0, stream>>>(deg, 1.f, NN / 4);
  degree_kernel<<<(E + 255) / 256, 256, 0, stream>>>(dst, deg, E);
  rsqrt_kernel<<<NN / 256, 256, 0, stream>>>(deg);

  // ---- segment matmul: h = x @ Wdyn[g]  (bf16 out) ----
  gemm_bf16_kernel<8, true><<<NN / 128, 256, 0, stream>>>(
      xb, wdyn, nullptr, hb, HH, SEGS);

  // ---- 3 GCN layers ----
  for (int l = 0; l < 3; ++l) {
    gemm_bf16_kernel<8, false><<<NN / 128, 256, 0, stream>>>(
        hb, cwt + (size_t)l * HH * HH, hw, nullptr, HH, NN);
    fill4_kernel<<<NN * HH / 4 / 256, 256, 0, stream>>>(accb, 0.f, NN * HH / 4);
    scatter_kernel<<<(E * 32 + 255) / 256, 256, 0, stream>>>(src, dst, deg, hw,
                                                             accb, E);
    finalize_kernel<<<NN * HH / 256, 256, 0, stream>>>(accb, hw, deg,
                                                       cb + l * HH, hb);
  }

  // ---- logits GEMM (padded N=48) + log_softmax ----
  gemm_bf16_kernel<3, false><<<NN / 128, 256, 0, stream>>>(hb, l1p, hw, nullptr,
                                                           NCLSP, NN);
  logsoftmax_kernel<<<NN / 256, 256, 0, stream>>>(hw, l1b, out);
}
